// Encoder_12266426597949
// MI455X (gfx1250) — matrix operand order repaired and checked by reading, not compiled, not run on hardware
//
#include <hip/hip_runtime.h>
#include <stdint.h>

// ---------------------------------------------------------------------------
// Bidirectional LSTM (B=32, T=512, F=64, U=512) for gfx1250 (MI455X).
// Persistent cooperative kernel: 2 directions x 16 workgroups; each WG owns
// 32 units (=128 gate columns), keeps its bf16 R-slice [128][520-padded]
// resident in LDS, stages h into (padded) LDS once per step, and runs the
// whole T=512 recurrence with v_wmma_f32_16x16x32_bf16.
// x@K for step t+1 is computed *before* the per-step grid barrier so it
// overlaps barrier wait; only h@R (all operands in conflict-free LDS) is on
// the sequential critical path.
// ---------------------------------------------------------------------------

typedef __bf16 bf16;
typedef __bf16 v16bf __attribute__((ext_vector_type(16)));
typedef __bf16 v8bf  __attribute__((ext_vector_type(8)));
typedef float  v8f   __attribute__((ext_vector_type(8)));

#define B_ 32
#define T_ 512
#define F_ 64
#define U_ 512
#define NG_ 2048            // 4*U
#define NWG_DIR 16          // workgroups per direction
#define UB 32               // units per workgroup
#define THREADS 128         // 4 waves of 32
#define RS 520              // padded LDS row stride in bf16 (1040B = 260 dwords
                            //  -> lane l starts at bank 4l mod 64: conflict-free)
#define RS4 65              // uint4 (16B) chunks per padded row

// LDS layout (dynamic):
//   rslice: [128][RS] bf16   = 133120 B
//   zbuf  : [4][32][32] f32  =  16384 B
//   hs    : [32][RS]  bf16   =  33280 B
#define LDS_ZOFF   133120
#define LDS_HOFF   149504
#define LDS_TOTAL  182784

__device__ __forceinline__ float sigmoidf_(float x) { return 1.0f / (1.0f + __expf(-x)); }
__device__ __forceinline__ float swishf_(float x)   { return x / (1.0f + __expf(-x)); }

// ---------------- prep kernels -------------------------------------------

__global__ void k_prep_x(const float* __restrict__ x, bf16* __restrict__ x16) {
    int i = blockIdx.x * blockDim.x + threadIdx.x;   // B*T*F = 1048576
    if (i < B_ * T_ * F_) x16[i] = (bf16)x[i];
}

// Ktp[d][n][k] = K_d[k][n], n in [0,2048), k in [0,64)
__global__ void k_pack_K(const float* __restrict__ Kf, const float* __restrict__ Kb,
                         bf16* __restrict__ Ktp) {
    int i = blockIdx.x * blockDim.x + threadIdx.x;   // 2*64*2048
    if (i >= 2 * F_ * NG_) return;
    int d = i / (F_ * NG_);
    int r = i % (F_ * NG_);
    int k = r / NG_;              // consecutive threads vary n -> coalesced reads
    int n = r % NG_;
    const float* src = d ? Kb : Kf;
    Ktp[(size_t)d * NG_ * F_ + (size_t)n * F_ + k] = (bf16)src[(size_t)k * NG_ + n];
}

// Rtp[d][n][k] = R_d[k][n], n in [0,2048), k in [0,512)
__global__ void k_pack_R(const float* __restrict__ Rf, const float* __restrict__ Rb,
                         bf16* __restrict__ Rtp) {
    int i = blockIdx.x * blockDim.x + threadIdx.x;   // 2*512*2048
    if (i >= 2 * U_ * NG_) return;
    int d = i / (U_ * NG_);
    int r = i % (U_ * NG_);
    int k = r / NG_;
    int n = r % NG_;
    const float* src = d ? Rb : Rf;
    Rtp[(size_t)d * NG_ * U_ + (size_t)n * U_ + k] = (bf16)src[(size_t)k * NG_ + n];
}

__global__ void k_init(bf16* __restrict__ hbuf, unsigned* __restrict__ bars) {
    int i = blockIdx.x * blockDim.x + threadIdx.x;
    if (i < 2 * 2 * B_ * U_) hbuf[i] = (bf16)0.0f;   // [dir][buf][B][U]
    if (i < 4) bars[i] = 0u;                          // [dir][{count,phase}]
}

// ---------------- main persistent kernel ---------------------------------

__global__ __launch_bounds__(THREADS, 1)
void lstm_persistent(const bf16* __restrict__ x16,     // [B][T][F] bf16
                     const bf16* __restrict__ Ktp,     // [2][2048][64]
                     const bf16* __restrict__ Rtp,     // [2][2048][512]
                     const float* __restrict__ bias_f, // [2048]
                     const float* __restrict__ bias_b, // [2048]
                     bf16* __restrict__ hbuf,          // [2][2][32][512]
                     unsigned* __restrict__ bars,      // [2][2]
                     float* __restrict__ out)          // [B][T][2U]
{
    extern __shared__ char smem[];
    bf16*  rslice = (bf16*)smem;                 // [128][RS]
    float* zbuf   = (float*)(smem + LDS_ZOFF);   // [4][32][32]
    bf16*  hs     = (bf16*)(smem + LDS_HOFF);    // [32][RS]

    const int dir  = blockIdx.x >> 4;
    const int wg   = blockIdx.x & 15;
    const int u0   = wg * UB;
    const int tid  = threadIdx.x;
    const int wave = tid >> 5;        // == gate index (i,f,g,o)
    const int lane = tid & 31;
    const int lh   = lane >> 4;       // lane half
    const int ll   = lane & 15;

    const bf16*  Rt   = Rtp + (size_t)dir * NG_ * U_;
    const bf16*  Kt   = Ktp + (size_t)dir * NG_ * F_;
    const float* bias = dir ? bias_b : bias_f;
    bf16* hdir = hbuf + (size_t)dir * 2 * B_ * U_;
    unsigned* bcount = bars + dir * 2;
    unsigned* bphase = bars + dir * 2 + 1;

    // ---- stage R slice into padded LDS: row n_local -> gate*512+u0+ulocal
    for (int i = tid; i < 128 * 64; i += THREADS) {   // 64 uint4 chunks per row
        int row = i >> 6, chunk = i & 63;
        int gate = row >> 5, ul = row & 31;
        int ng = gate * U_ + u0 + ul;
        ((uint4*)rslice)[row * RS4 + chunk] =
            ((const uint4*)(Rt + (size_t)ng * U_))[chunk];
    }

    // ---- per-thread elementwise state: 8 (b,u) pairs, b fixed per thread
    const int brow = tid >> 2;                 // batch row 0..31
    float cst[8], bi[8], bff[8], bg[8], bo[8];
#pragma unroll
    for (int j = 0; j < 8; ++j) {
        int ul = (tid & 3) * 8 + j;
        cst[j] = 0.0f;
        bi[j]  = bias[0 * U_ + u0 + ul];
        bff[j] = bias[1 * U_ + u0 + ul];
        bg[j]  = bias[2 * U_ + u0 + ul];
        bo[j]  = bias[3 * U_ + u0 + ul];
    }

    const int nt0 = wave * U_ + u0;            // B-frag global col base (Ktp)
    v8f a00, a01, a10, a11;

    // x_t @ K for a given timestep, accumulated into a00..a11 (no h dep).
    auto xgemm = [&](int t, bool init) {
#pragma unroll
        for (int k0 = 0; k0 < F_; k0 += 32) {
            union { v16bf v; v8bf h2[2]; } fa0, fa1;
            const bf16* xr0 = x16 + ((size_t)ll * T_ + t) * F_;
            const bf16* xr1 = x16 + ((size_t)(16 + ll) * T_ + t) * F_;
            fa0.h2[0] = *(const v8bf*)(xr0 + k0 + 16 * lh);
            fa0.h2[1] = *(const v8bf*)(xr0 + k0 + 16 * lh + 8);
            fa1.h2[0] = *(const v8bf*)(xr1 + k0 + 16 * lh);
            fa1.h2[1] = *(const v8bf*)(xr1 + k0 + 16 * lh + 8);
            v16bf fb0 = *(const v16bf*)(Kt + (size_t)(nt0 + ll)      * F_ + k0 + 16 * lh);
            v16bf fb1 = *(const v16bf*)(Kt + (size_t)(nt0 + 16 + ll) * F_ + k0 + 16 * lh);
            if (init && k0 == 0) {
                v8f z = {0};
                a00 = __builtin_amdgcn_wmma_f32_16x16x32_bf16(false, fa0.v, false, fb0, (short)0, z, false, false);
                a01 = __builtin_amdgcn_wmma_f32_16x16x32_bf16(false, fa0.v, false, fb1, (short)0, z, false, false);
                a10 = __builtin_amdgcn_wmma_f32_16x16x32_bf16(false, fa1.v, false, fb0, (short)0, z, false, false);
                a11 = __builtin_amdgcn_wmma_f32_16x16x32_bf16(false, fa1.v, false, fb1, (short)0, z, false, false);
            } else {
                a00 = __builtin_amdgcn_wmma_f32_16x16x32_bf16(false, fa0.v, false, fb0, (short)0, a00, false, false);
                a01 = __builtin_amdgcn_wmma_f32_16x16x32_bf16(false, fa0.v, false, fb1, (short)0, a01, false, false);
                a10 = __builtin_amdgcn_wmma_f32_16x16x32_bf16(false, fa1.v, false, fb0, (short)0, a10, false, false);
                a11 = __builtin_amdgcn_wmma_f32_16x16x32_bf16(false, fa1.v, false, fb1, (short)0, a11, false, false);
            }
        }
    };

    // accumulators for t=0 computed up-front (h is zero-initialized in ws)
    xgemm(dir ? (T_ - 1) : 0, true);

    for (int ti = 0; ti < T_; ++ti) {
        const int t = dir ? (T_ - 1 - ti) : ti;
        const bf16* hin  = hdir + (size_t)(ti & 1) * B_ * U_;
        bf16*       hout = hdir + (size_t)((ti + 1) & 1) * B_ * U_;

        // ---- stage h into padded LDS (32KB, conflict-free rows)
        for (int i = tid; i < 32 * 64; i += THREADS) {
            int row = i >> 6, chunk = i & 63;
            ((uint4*)hs)[row * RS4 + chunk] =
                ((const uint4*)(hin + (size_t)row * U_))[chunk];
        }
        __syncthreads();

        // ---- recurrent contribution: h @ R (all operands in LDS)
#pragma unroll 4
        for (int k0 = 0; k0 < U_; k0 += 32) {
            union { v16bf v; v8bf h2[2]; } fa0, fa1, fb0, fb1;
            const bf16* hr0 = hs + (size_t)ll * RS;
            const bf16* hr1 = hs + (size_t)(16 + ll) * RS;
            fa0.h2[0] = *(const v8bf*)(hr0 + k0 + 16 * lh);
            fa0.h2[1] = *(const v8bf*)(hr0 + k0 + 16 * lh + 8);
            fa1.h2[0] = *(const v8bf*)(hr1 + k0 + 16 * lh);
            fa1.h2[1] = *(const v8bf*)(hr1 + k0 + 16 * lh + 8);
            const bf16* rb0 = rslice + (size_t)(wave * 32 + ll)      * RS + k0 + 16 * lh;
            const bf16* rb1 = rslice + (size_t)(wave * 32 + 16 + ll) * RS + k0 + 16 * lh;
            fb0.h2[0] = *(const v8bf*)(rb0);
            fb0.h2[1] = *(const v8bf*)(rb0 + 8);
            fb1.h2[0] = *(const v8bf*)(rb1);
            fb1.h2[1] = *(const v8bf*)(rb1 + 8);
            a00 = __builtin_amdgcn_wmma_f32_16x16x32_bf16(false, fa0.v, false, fb0.v, (short)0, a00, false, false);
            a01 = __builtin_amdgcn_wmma_f32_16x16x32_bf16(false, fa0.v, false, fb1.v, (short)0, a01, false, false);
            a10 = __builtin_amdgcn_wmma_f32_16x16x32_bf16(false, fa1.v, false, fb0.v, (short)0, a10, false, false);
            a11 = __builtin_amdgcn_wmma_f32_16x16x32_bf16(false, fa1.v, false, fb1.v, (short)0, a11, false, false);
        }

        // ---- spill z tiles to LDS (C layout: VGPR r -> M=r+8*lh, N=ll)
        float* zg = zbuf + wave * (B_ * UB);
#pragma unroll
        for (int r = 0; r < 8; ++r) {
            int m = r + 8 * lh;
            zg[m * UB + ll]             = a00[r];
            zg[m * UB + 16 + ll]        = a01[r];
            zg[(16 + m) * UB + ll]      = a10[r];
            zg[(16 + m) * UB + 16 + ll] = a11[r];
        }
        __syncthreads();

        // ---- gate nonlinearities + state update (f32)
#pragma unroll
        for (int j = 0; j < 8; ++j) {
            int ul = (tid & 3) * 8 + j;
            float iv = sigmoidf_(zbuf[0 * 1024 + brow * UB + ul] + bi[j]);
            float fv = sigmoidf_(zbuf[1 * 1024 + brow * UB + ul] + bff[j]);
            float gv = swishf_  (zbuf[2 * 1024 + brow * UB + ul] + bg[j]);
            float ov = sigmoidf_(zbuf[3 * 1024 + brow * UB + ul] + bo[j]);
            float cn = fv * cst[j] + iv * gv;
            cst[j] = cn;
            float hv = ov * swishf_(cn);
            hout[(size_t)brow * U_ + u0 + ul] = (bf16)hv;
            out[(size_t)brow * (T_ * 2 * U_) + (size_t)t * (2 * U_) + dir * U_ + u0 + ul] = hv;
        }

        // ---- overlap: next step's x@K (independent of h) before the barrier
        if (ti + 1 < T_) xgemm(dir ? (T_ - 2 - ti) : (ti + 1), true);

        // ---- per-direction grid barrier (16 WGs), sense-reversing phase
        __threadfence();
        __syncthreads();
        if (tid == 0) {
            volatile unsigned* vph = (volatile unsigned*)bphase;
            unsigned ph = *vph;
            unsigned old = atomicAdd(bcount, 1u);
            if (old == NWG_DIR - 1) {
                atomicExch(bcount, 0u);
                __threadfence();
                atomicAdd(bphase, 1u);
            } else {
                while (*vph == ph) { __builtin_amdgcn_s_sleep(1); }
            }
            __threadfence();
        }
        __syncthreads();
    }
}

// ---------------- launcher ------------------------------------------------

extern "C" void kernel_launch(void* const* d_in, const int* in_sizes, int n_in,
                              void* d_out, int out_size, void* d_ws, size_t ws_size,
                              hipStream_t stream) {
    const float* x  = (const float*)d_in[0];
    const float* Kf = (const float*)d_in[1];
    const float* Rf = (const float*)d_in[2];
    const float* bf = (const float*)d_in[3];
    const float* Kb = (const float*)d_in[4];
    const float* Rb = (const float*)d_in[5];
    const float* bb = (const float*)d_in[6];
    float* out = (float*)d_out;

    // workspace layout (bytes), ~6.95 MB total
    char* ws = (char*)d_ws;
    bf16*     x16  = (bf16*)(ws);                       // 2,097,152
    bf16*     Ktp  = (bf16*)(ws + 2097152);             //   524,288
    bf16*     Rtp  = (bf16*)(ws + 2621440);             // 4,194,304
    bf16*     hbuf = (bf16*)(ws + 6815744);             //   131,072
    unsigned* bars = (unsigned*)(ws + 6946816);         //        16

    k_prep_x<<<(B_ * T_ * F_ + 255) / 256, 256, 0, stream>>>(x, x16);
    k_pack_K<<<(2 * F_ * NG_ + 255) / 256, 256, 0, stream>>>(Kf, Kb, Ktp);
    k_pack_R<<<(2 * U_ * NG_ + 255) / 256, 256, 0, stream>>>(Rf, Rb, Rtp);
    k_init  <<<(2 * 2 * B_ * U_ + 255) / 256, 256, 0, stream>>>(hbuf, bars);

    const size_t shmem = LDS_TOTAL;                     // 182784 B
    (void)hipFuncSetAttribute((const void*)lstm_persistent,
                              hipFuncAttributeMaxDynamicSharedMemorySize,
                              (int)shmem);
    lstm_persistent<<<dim3(2 * NWG_DIR), dim3(THREADS), shmem, stream>>>(
        x16, Ktp, Rtp, bf, bb, hbuf, bars, out);
}
